// EuclideanCodebook_64278480552680
// MI455X (gfx1250) — matrix operand order repaired and checked by reading, not compiled, not run on hardware
//
#include <hip/hip_runtime.h>

// ---------------------------------------------------------------------------
// Euclidean codebook (VQ-VAE EMA update) for MI455X / gfx1250.
//
// Dominant work: dist argmin needs xf[65536x128] @ embed^T[128x1024].
// That 17.2 GFLOP GEMM runs on v_wmma_f32_16x16x32_f16 (f16 in, f32 acc).
// ||x||^2 is constant per row -> dropped from the argmin.
//
// v2: each wave now owns a 32-row M tile (two A-fragment sets, two
// accumulators), so every B fragment (codebook tile) feeds TWO WMMAs.
// This halves the B-side load traffic per FLOP that dominated the v1
// inner loop (s_wait_loadcnt stalls between WMMAs).
// ---------------------------------------------------------------------------

#define DIM   128
#define KCODES 1024
#define DECAY 0.99f
#define OMD   0.01f     // 1 - DECAY
#define EPS   1e-6f

typedef _Float16 half_t;
typedef __attribute__((ext_vector_type(16))) _Float16 v16h;
typedef __attribute__((ext_vector_type(8)))  float    v8f;

union Frag16 {            // 16 halfs = 32 bytes = two b128 loads
  v16h  h;
  uint4 q[2];
};
union Pack4 { half_t h[4]; uint2 u; };

// ---------------------------------------------------------------------------
// Kernel 1: embed f32 -> f16 copy, ||e_k||^2, zero scatter accumulators.
// grid = KCODES blocks of 128 threads (one code per block).
// ---------------------------------------------------------------------------
__global__ __launch_bounds__(128) void vq_prep_kernel(
    const float* __restrict__ embed,
    half_t* __restrict__ embed_h,
    float* __restrict__ enorm,
    float* __restrict__ cnt,
    float* __restrict__ esum)
{
  const int k = blockIdx.x;
  const int t = threadIdx.x;            // 0..127 == d index
  const float v = embed[(size_t)k * DIM + t];
  embed_h[(size_t)k * DIM + t] = (half_t)v;
  esum[(size_t)k * DIM + t] = 0.0f;
  if (t == 0) cnt[k] = 0.0f;

  __shared__ float s[128];
  s[t] = v * v;
  __syncthreads();
  for (int off = 64; off > 0; off >>= 1) {
    if (t < off) s[t] += s[t + off];
    __syncthreads();
  }
  if (t == 0) enorm[k] = s[0];
}

// ---------------------------------------------------------------------------
// Kernel 2: main VQ kernel.  128 threads = 4 waves; each wave owns a 32-row
// tile of x (two 16-row WMMA M-tiles sharing each B fragment), scans all
// 1024 codes with WMMA, then emits indices, the quantize gather, and the
// scatter-sum (counts + x rows) via f32 atomics.
//
// WMMA fragment layouts (CDNA5 ISA 7.12.2, wave32):
//   A (16x32 f16):  lane l -> row M=l&15, half h=l>>4 supplies
//                   K = h*8..h*8+7  and  K = 16+h*8..16+h*8+7
//                   -> two 16-byte contiguous chunks of a row-major tile.
//   B (32x16 f16):  lane l -> column N=l&15, K = h*8..h*8+7 and 16+h*8..16+h*8+7
//                   -> two 16-byte chunks of a row-major embed row.
//   C/D (16x16 f32): VGPR r: lanes 0-15 -> M=r, N=lane;
//                            lanes 16-31 -> M=8+r, N=lane-16.
// ---------------------------------------------------------------------------
__global__ __launch_bounds__(128) void vq_main_kernel(
    const float*  __restrict__ x,        // [N][DIM] f32
    const float*  __restrict__ embed,    // [K][DIM] f32 (for gather)
    const half_t* __restrict__ embed_h,  // [K][DIM] f16
    const float*  __restrict__ enorm,    // [K]
    float* __restrict__ cnt,             // [K]   scatter counts
    float* __restrict__ esum,            // [K][DIM] scatter sums
    float* __restrict__ out_q,           // [N][DIM] quantize
    float* __restrict__ out_ind)         // [N] indices as float
{
  __shared__ __align__(16) half_t xh[4][32 * DIM];  // 4 waves * 8KB = 32KB
  __shared__ int rowIdx[4][32];

  const int tid  = threadIdx.x;
  const int w    = tid >> 5;          // wave in block
  const int lane = tid & 31;
  const int col  = lane & 15;         // N column inside a 16-wide tile
  const int hi   = lane >> 4;         // half-wave select
  const int rowBase = blockIdx.x * 128 + w * 32;

  // ---- stage 32 rows of x into LDS as f16 (row-major, stride DIM) ----
  for (int r = 0; r < 32; ++r) {
    const float4 f = *(const float4*)(x + (size_t)(rowBase + r) * DIM + lane * 4);
    Pack4 p;
    p.h[0] = (half_t)f.x; p.h[1] = (half_t)f.y;
    p.h[2] = (half_t)f.z; p.h[3] = (half_t)f.w;
    *(uint2*)&xh[w][r * DIM + lane * 4] = p.u;
  }
  __syncthreads();

  // ---- build the A-fragments for both 16-row M tiles (K = 0..127) ----
  Frag16 a0[4], a1[4];
#pragma unroll
  for (int kb = 0; kb < 4; ++kb) {
    const half_t* b0 = &xh[w][col * DIM + kb * 32];
    a0[kb].q[0] = *(const uint4*)(b0 + hi * 8);
    a0[kb].q[1] = *(const uint4*)(b0 + 16 + hi * 8);
    const half_t* b1 = &xh[w][(16 + col) * DIM + kb * 32];
    a1[kb].q[0] = *(const uint4*)(b1 + hi * 8);
    a1[kb].q[1] = *(const uint4*)(b1 + 16 + hi * 8);
  }

  // ---- scan all code tiles, track per-lane running (min, argmin) ----
  float minv0[8], minv1[8];
  int   mini0[8], mini1[8];
#pragma unroll
  for (int r = 0; r < 8; ++r) {
    minv0[r] = 3.4e38f; mini0[r] = 0;
    minv1[r] = 3.4e38f; mini1[r] = 0;
  }

  for (int nt = 0; nt < KCODES / 16; ++nt) {
    const int code = nt * 16 + col;
    const half_t* brow = embed_h + (size_t)code * DIM;
    if (nt + 1 < KCODES / 16)   // gfx1250 global_prefetch of next code tile
      __builtin_prefetch(brow + 16 * DIM, 0, 1);

    v8f acc0 = {};
    v8f acc1 = {};
#pragma unroll
    for (int kb = 0; kb < 4; ++kb) {
      Frag16 b;
      const half_t* bp = brow + kb * 32 + hi * 8;
      b.q[0] = *(const uint4*)(bp);
      b.q[1] = *(const uint4*)(bp + 16);
      acc0 = __builtin_amdgcn_wmma_f32_16x16x32_f16(
          false, a0[kb].h, false, b.h, (short)0, acc0, false, false);
      acc1 = __builtin_amdgcn_wmma_f32_16x16x32_f16(
          false, a1[kb].h, false, b.h, (short)0, acc1, false, false);
    }

    const float en = enorm[code];   // dist(k) = ||e||^2 - 2 x.e  (+const)
#pragma unroll
    for (int r = 0; r < 8; ++r) {
      const float d0 = en - 2.0f * acc0[r];
      if (d0 < minv0[r]) { minv0[r] = d0; mini0[r] = code; }
      const float d1 = en - 2.0f * acc1[r];
      if (d1 < minv1[r]) { minv1[r] = d1; mini1[r] = code; }
    }
  }

  // ---- butterfly min-reduce over the 16-lane group (xor 1,2,4,8) ----
#pragma unroll
  for (int r = 0; r < 8; ++r) {
    {
      float v = minv0[r]; int i = mini0[r];
#pragma unroll
      for (int m = 8; m >= 1; m >>= 1) {
        const float ov = __shfl_xor(v, m, 32);
        const int   oi = __shfl_xor(i, m, 32);
        if (ov < v || (ov == v && oi < i)) { v = ov; i = oi; }  // first-idx ties
      }
      minv0[r] = v; mini0[r] = i;
    }
    {
      float v = minv1[r]; int i = mini1[r];
#pragma unroll
      for (int m = 8; m >= 1; m >>= 1) {
        const float ov = __shfl_xor(v, m, 32);
        const int   oi = __shfl_xor(i, m, 32);
        if (ov < v || (ov == v && oi < i)) { v = ov; i = oi; }
      }
      minv1[r] = v; mini1[r] = i;
    }
  }

  // lanes 0 and 16 hold rows M = r and M = 8+r respectively (per M tile)
  if (col == 0) {
#pragma unroll
    for (int r = 0; r < 8; ++r) {
      {
        const int m   = hi * 8 + r;         // rows 0..15
        const int row = rowBase + m;
        rowIdx[w][m]  = mini0[r];
        out_ind[row]  = (float)mini0[r];
        atomicAdd(cnt + mini0[r], 1.0f);    // one_hot segment_sum
      }
      {
        const int m   = 16 + hi * 8 + r;    // rows 16..31
        const int row = rowBase + m;
        rowIdx[w][m]  = mini1[r];
        out_ind[row]  = (float)mini1[r];
        atomicAdd(cnt + mini1[r], 1.0f);
      }
    }
  }
  __syncthreads();

  // ---- quantize gather + embed_sum scatter (f32 atomics) ----
#pragma unroll 1
  for (int m = 0; m < 32; ++m) {
    const int row = rowBase + m;
    const int idx = rowIdx[w][m];
    const float4 e4 = *(const float4*)(embed + (size_t)idx * DIM + lane * 4);
    *(float4*)(out_q + (size_t)row * DIM + lane * 4) = e4;

    const float4 xv = *(const float4*)(x + (size_t)row * DIM + lane * 4);
    float* dst = esum + (size_t)idx * DIM + lane * 4;
    atomicAdd(dst + 0, xv.x);
    atomicAdd(dst + 1, xv.y);
    atomicAdd(dst + 2, xv.z);
    atomicAdd(dst + 3, xv.w);
  }
}

// ---------------------------------------------------------------------------
// Kernel 3: cluster_size EMA + total reduction (single 1024-thread block).
// ---------------------------------------------------------------------------
__global__ __launch_bounds__(1024) void vq_cs_kernel(
    const float* __restrict__ cs,
    const float* __restrict__ cnt,
    float* __restrict__ out_cs,
    float* __restrict__ total_ws)
{
  __shared__ float s[KCODES];
  const int t = threadIdx.x;
  const float v = cs[t] * DECAY + cnt[t] * OMD;
  out_cs[t] = v;
  s[t] = v;
  __syncthreads();
  for (int off = 512; off > 0; off >>= 1) {
    if (t < off) s[t] += s[t + off];
    __syncthreads();
  }
  if (t == 0) *total_ws = s[0];
}

// ---------------------------------------------------------------------------
// Kernel 4: embed_avg EMA + smoothed normalization.
// ---------------------------------------------------------------------------
__global__ __launch_bounds__(128) void vq_embed_kernel(
    const float* __restrict__ ea,
    const float* __restrict__ esum,
    const float* __restrict__ out_cs,
    const float* __restrict__ total_ws,
    float* __restrict__ out_ea,
    float* __restrict__ out_embed)
{
  const int k = blockIdx.x;
  const int d = threadIdx.x;
  const size_t off = (size_t)k * DIM + d;
  const float ean = ea[off] * DECAY + esum[off] * OMD;
  out_ea[off] = ean;
  const float total    = *total_ws;
  const float csn      = out_cs[k];
  const float smoothed = (csn + EPS) / (total + EPS * (float)KCODES) * total;
  out_embed[off] = ean / smoothed;
}

// ---------------------------------------------------------------------------
// Launcher.  Inputs: x[64,1024,128], embed[1024,128], cluster_size[1024],
// embed_avg[1024,128].  Outputs concatenated flat:
//   quantize[N*128] | embed_ind[N] | cluster_size_new[1024]
//   | embed_avg_new[1024*128] | embed_new[1024*128]
// ---------------------------------------------------------------------------
extern "C" void kernel_launch(void* const* d_in, const int* in_sizes, int n_in,
                              void* d_out, int out_size, void* d_ws, size_t ws_size,
                              hipStream_t stream) {
  (void)n_in; (void)out_size; (void)ws_size;
  const float* x     = (const float*)d_in[0];
  const float* embed = (const float*)d_in[1];
  const float* cs    = (const float*)d_in[2];
  const float* ea    = (const float*)d_in[3];
  const int N = in_sizes[0] / DIM;               // 65536

  float* out       = (float*)d_out;
  float* out_q     = out;                         // [N][128]
  float* out_ind   = out_q  + (size_t)N * DIM;    // [N]
  float* out_cs    = out_ind + N;                 // [1024]
  float* out_ea    = out_cs + KCODES;             // [1024][128]
  float* out_embed = out_ea + (size_t)KCODES * DIM;

  // workspace layout (bytes)
  char* ws = (char*)d_ws;
  half_t* embed_h = (half_t*)(ws);                           // 256 KB
  float*  enorm   = (float*)(ws + 262144);                   //   4 KB
  float*  cnt     = (float*)(ws + 266240);                   //   4 KB
  float*  esum    = (float*)(ws + 270336);                   // 512 KB
  float*  total   = (float*)(ws + 794624);                   //   4 B

  vq_prep_kernel<<<KCODES, 128, 0, stream>>>(embed, embed_h, enorm, cnt, esum);
  vq_main_kernel<<<N / 128, 128, 0, stream>>>(x, embed, embed_h, enorm,
                                              cnt, esum, out_q, out_ind);
  vq_cs_kernel<<<1, 1024, 0, stream>>>(cs, cnt, out_cs, total);
  vq_embed_kernel<<<KCODES, 128, 0, stream>>>(ea, esum, out_cs, total,
                                              out_ea, out_embed);
}
